// Transformer_47588237640422
// MI455X (gfx1250) — matrix operand order repaired and checked by reading, not compiled, not run on hardware
//
#include <hip/hip_runtime.h>
#include <hip/hip_bf16.h>
#include <stdint.h>

// ---------------------------------------------------------------------------
// Model dims
// ---------------------------------------------------------------------------
#define LAYERS 6
#define NHEAD  6
#define DMODEL 768
#define HSZ    128
#define FFDIM  3072
#define TCTX   1024
#define BATCH  4
#define VOCAB  32000
#define BT     (BATCH * TCTX)

typedef _Float16 v16h __attribute__((ext_vector_type(16)));
typedef float    v8f  __attribute__((ext_vector_type(8)));
typedef _Float16 h2t  __attribute__((ext_vector_type(2)));
typedef unsigned int u32x4 __attribute__((ext_vector_type(4)));
typedef int          i32x8 __attribute__((ext_vector_type(8)));
typedef int          i32x4 __attribute__((ext_vector_type(4)));

#if defined(__has_builtin)
#if __has_builtin(__builtin_amdgcn_tensor_load_to_lds) && \
    __has_builtin(__builtin_amdgcn_s_wait_tensorcnt)
#define HAVE_TDM 1
#endif
#if __has_builtin(__builtin_amdgcn_sched_barrier)
#define SCHED_FENCE() __builtin_amdgcn_sched_barrier(0)
#endif
#endif
#ifndef SCHED_FENCE
#define SCHED_FENCE()
#endif

// ---------------------------------------------------------------------------
// TDM: DMA a [rows x width] f16 tile (row stride = strideElems) into LDS.
// ---------------------------------------------------------------------------
#ifdef HAVE_TDM
__device__ __forceinline__ void tdm_load_tile_f16(const _Float16* gsrc,
                                                  uint32_t ldsByteAddr,
                                                  int rows, int width,
                                                  int strideElems) {
  uint64_t ga = (uint64_t)(uintptr_t)gsrc;
  u32x4 g0 = {1u,                                  // count=1 (valid user D#)
              ldsByteAddr,                         // lds_addr
              (uint32_t)ga,                        // global_addr[31:0]
              (uint32_t)((ga >> 32) & 0x1FFFFFFu) | (2u << 30)}; // [56:32]|type=2
  uint32_t w = (uint32_t)width, r = (uint32_t)rows, s = (uint32_t)strideElems;
  i32x8 g1 = {(int)(1u << 16),                     // data_size=1 -> 2 bytes
              (int)((w & 0xFFFFu) << 16),          // tensor_dim0[15:0] @ [63:48]
              (int)((w >> 16) | ((r & 0xFFFFu) << 16)),   // dim0 hi | dim1 lo
              (int)((r >> 16) | (w << 16)),        // dim1 hi | tile_dim0
              (int)r,                              // tile_dim1 | tile_dim2=0
              (int)s,                              // dim0_stride[31:0]
              0, 0};
  i32x4 z4 = {0, 0, 0, 0};
#if __clang_major__ >= 23
  i32x8 z8 = {0, 0, 0, 0, 0, 0, 0, 0};
  __builtin_amdgcn_tensor_load_to_lds(g0, g1, z4, z4, z8, 0);
#else
  __builtin_amdgcn_tensor_load_to_lds(g0, g1, z4, z4, 0);
#endif
}
#endif

// ---------------------------------------------------------------------------
// Weight prep: all weights stored f16 *transposed* ([N][K] row-major) so
// GEMM B tiles are plain strided tiles -> TDM-stageable, no LDS transpose.
// ---------------------------------------------------------------------------
__global__ void cvt_f32_to_f16_kernel(const float* __restrict__ src,
                                      _Float16* __restrict__ dst, int n) {
  int i = blockIdx.x * blockDim.x + threadIdx.x;
  if (i < n) dst[i] = (_Float16)src[i];
}

// wq/wk/wv [L,H,D,HS] f32 -> [L, H*HS, D] f16 (transposed, head-major rows)
__global__ void cvt_qkv_kernel(const float* __restrict__ src,
                               _Float16* __restrict__ dst) {
  int i = blockIdx.x * blockDim.x + threadIdx.x;  // over L*H*D*HS
  int e = i % HSZ;
  int r = i / HSZ;
  int d = r % DMODEL;
  int r2 = r / DMODEL;
  int h = r2 % NHEAD;
  int l = r2 / NHEAD;
  dst[l * DMODEL * DMODEL + (h * HSZ + e) * DMODEL + d] = (_Float16)src[i];
}

// w [L,K,N] f32 -> [L,N,K] f16
__global__ void cvt_w_transpose_kernel(const float* __restrict__ src,
                                       _Float16* __restrict__ dst,
                                       int Kd, int Nd) {
  int i = blockIdx.x * blockDim.x + threadIdx.x;  // over L*K*N
  int kn = Kd * Nd;
  int l = i / kn;
  int r = i % kn;
  int k = r / Nd;
  int n = r % Nd;
  dst[l * kn + n * Kd + k] = (_Float16)src[i];
}

// V [B*T, D] -> VT [B, H, HS, T]
__global__ void transpose_v_kernel(const _Float16* __restrict__ v,
                                   _Float16* __restrict__ vt) {
  int i = blockIdx.x * blockDim.x + threadIdx.x;  // over BT*DMODEL
  int d = i % DMODEL;
  int row = i / DMODEL;
  int b = row / TCTX, t = row % TCTX;
  int h = d / HSZ, e = d % HSZ;
  vt[((b * NHEAD + h) * HSZ + e) * TCTX + t] = v[i];
}

__global__ void embed_kernel(const int* __restrict__ idx,
                             const float* __restrict__ tok,
                             const float* __restrict__ pos,
                             float* __restrict__ x) {
  int row = blockIdx.x;
  int t = row % TCTX;
  int token = idx[row];
  for (int i = threadIdx.x; i < DMODEL; i += blockDim.x)
    x[row * DMODEL + i] = tok[token * DMODEL + i] + pos[t * DMODEL + i];
}

__global__ __launch_bounds__(256)
void layernorm_kernel(const float* __restrict__ x,
                      const float* __restrict__ g,
                      const float* __restrict__ b,
                      _Float16* __restrict__ out) {
  __shared__ float r1[256];
  __shared__ float r2[256];
  int row = blockIdx.x;
  int tid = threadIdx.x;
  const float* xr = x + row * DMODEL;
  float s = 0.f, s2 = 0.f;
  for (int i = tid; i < DMODEL; i += 256) {
    float v = xr[i];
    s += v; s2 += v * v;
  }
  r1[tid] = s; r2[tid] = s2;
  __syncthreads();
  for (int off = 128; off > 0; off >>= 1) {
    if (tid < off) { r1[tid] += r1[tid + off]; r2[tid] += r2[tid + off]; }
    __syncthreads();
  }
  float mean = r1[0] * (1.0f / DMODEL);
  float var  = r2[0] * (1.0f / DMODEL) - mean * mean;
  float inv  = rsqrtf(var + 1e-5f);
  _Float16* outr = out + row * DMODEL;
  for (int i = tid; i < DMODEL; i += 256)
    outr[i] = (_Float16)((xr[i] - mean) * inv * g[i] + b[i]);
}

// ---------------------------------------------------------------------------
// WMMA GEMM: C[M,N] = act(A[M,K] @ Bw[N,K]^T + bias + resid).
// 256 threads / 8 waves, 128x128 tile, K chunk 32, double-buffered LDS.
// Both tiles DMA'd by the Tensor Data Mover (wave0: A, wave1: B).
// sched fence splits "load all fragments" from "8 streaming WMMAs".
// ---------------------------------------------------------------------------
__global__ __launch_bounds__(256)
void gemm_wmma_kernel(const _Float16* __restrict__ A,
                      const _Float16* __restrict__ Bw,
                      const float* __restrict__ bias,
                      const float* __restrict__ resid,
                      float* __restrict__ Cf,
                      _Float16* __restrict__ Ch,
                      int M, int N, int K, int relu) {
  __shared__ _Float16 As[2][128 * 32];   // [m][k]   2 x 8 KB
  __shared__ _Float16 Bs[2][128 * 32];   // [n][k]   2 x 8 KB
  const int tid   = threadIdx.x;
  const int lane  = tid & 31;
  const int wave  = tid >> 5;
  const int mloc  = lane & 15;
  const int khalf = lane >> 4;
  const int rowBase = blockIdx.y * 128;
  const int colBase = blockIdx.x * 128;
  const int nk = K >> 5;

  const v8f vzero = {0.f, 0.f, 0.f, 0.f, 0.f, 0.f, 0.f, 0.f};
  v8f acc[8];
#pragma unroll
  for (int i = 0; i < 8; ++i) acc[i] = vzero;

#ifndef HAVE_TDM
  uint2 areg[4], brg[4];
  auto loadTiles = [&](int k0) {
#pragma unroll
    for (int i = 0; i < 4; ++i) {
      int id = tid + i * 256;          // 1024 uint2 chunks per tile
      int r  = id >> 3;
      int c  = (id & 7) << 2;
      areg[i] = *(const uint2*)(&A[(rowBase + r) * K + k0 + c]);
      brg[i]  = *(const uint2*)(&Bw[(colBase + r) * K + k0 + c]);
    }
  };
  auto storeTiles = [&](int buf) {
#pragma unroll
    for (int i = 0; i < 4; ++i) {
      int id = tid + i * 256;
      int r  = id >> 3;
      int c  = (id & 7) << 2;
      *(uint2*)(&As[buf][r * 32 + c]) = areg[i];
      *(uint2*)(&Bs[buf][r * 32 + c]) = brg[i];
    }
  };
#endif

  auto stage_issue = [&](int k0, int buf) {
#ifdef HAVE_TDM
    if (wave == 0)
      tdm_load_tile_f16(A + rowBase * K + k0,
                        (uint32_t)(uintptr_t)(&As[buf][0]), 128, 32, K);
    else if (wave == 1)
      tdm_load_tile_f16(Bw + colBase * K + k0,
                        (uint32_t)(uintptr_t)(&Bs[buf][0]), 128, 32, K);
#else
    loadTiles(k0);
#endif
  };
  auto stage_finish = [&](int buf) {
#ifdef HAVE_TDM
    if (wave < 2) __builtin_amdgcn_s_wait_tensorcnt(0);
    (void)buf;
#else
    storeTiles(buf);
#endif
  };

  stage_issue(0, 0);
  stage_finish(0);
  __syncthreads();

  for (int kc = 0; kc < nk; ++kc) {
    const int cur = kc & 1, nxt = cur ^ 1;
    const bool hasNext = (kc + 1 < nk);
    if (hasNext) {
      stage_issue((kc + 1) << 5, nxt);   // TDM streams while we compute
      __builtin_prefetch(&Bw[(colBase + (tid >> 1)) * K +
                             ((kc + 2 < nk ? kc + 2 : kc) << 5)], 0, 1);
    }

    // A fragment (16x32 f16, ISA lane layout)
    v16h af;
    h2t* ap = reinterpret_cast<h2t*>(&af);
#pragma unroll
    for (int v = 0; v < 8; ++v) {
      int kk = ((v < 4) ? (2 * v) : (16 + 2 * (v - 4))) + khalf * 8;
      ap[v] = *(const h2t*)(&As[cur][(wave * 16 + mloc) * 32 + kk]);
    }
    // All 8 B fragments loaded before any WMMA issues (sched fence below)
    v16h bfr[8];
#pragma unroll
    for (int ct = 0; ct < 8; ++ct) {
      h2t* bp = reinterpret_cast<h2t*>(&bfr[ct]);
#pragma unroll
      for (int v = 0; v < 8; ++v) {
        int kk = ((v < 4) ? (2 * v) : (16 + 2 * (v - 4))) + khalf * 8;
        bp[v] = *(const h2t*)(&Bs[cur][(ct * 16 + mloc) * 32 + kk]);
      }
    }
    SCHED_FENCE();
#pragma unroll
    for (int ct = 0; ct < 8; ++ct)
      acc[ct] = __builtin_amdgcn_wmma_f32_16x16x32_f16(
          false, af, false, bfr[ct], (short)0, acc[ct], false, false);
    SCHED_FENCE();

    if (hasNext) stage_finish(nxt);
    __syncthreads();
  }

#pragma unroll
  for (int ct = 0; ct < 8; ++ct) {
#pragma unroll
    for (int r = 0; r < 8; ++r) {
      int m = rowBase + wave * 16 + khalf * 8 + r;
      int n = colBase + ct * 16 + mloc;
      float v = acc[ct][r];
      if (bias)  v += bias[n];
      if (resid) v += resid[m * N + n];
      if (relu)  v = fmaxf(v, 0.f);
      if (Cf) Cf[m * N + n] = v;
      if (Ch) Ch[m * N + n] = (_Float16)v;
    }
  }
}

// ---------------------------------------------------------------------------
// Causal attention for one (b, h, 16-query-row block). 128 threads / 4 waves.
// ---------------------------------------------------------------------------
__global__ __launch_bounds__(128)
void attn_kernel(const _Float16* __restrict__ Q,
                 const _Float16* __restrict__ Kg,
                 const _Float16* __restrict__ VT,
                 _Float16* __restrict__ O) {
  __shared__ float sc[16 * TCTX];     // 64 KB score buffer
  const int tid   = threadIdx.x;
  const int lane  = tid & 31;
  const int wave  = tid >> 5;
  const int mloc  = lane & 15;
  const int khalf = lane >> 4;
  const int bid = blockIdx.x;
  const int qb  = bid & 63;
  const int h   = (bid >> 6) % NHEAD;
  const int b   = bid / (NHEAD * 64);
  const int headCol = h * HSZ;
  const int rowQ = b * TCTX + qb * 16 + mloc;
  const int vtBase = (b * NHEAD + h) * HSZ * TCTX;
  const float scale = 0.08838834764831845f;  // 1/sqrt(128)
  const v8f vzero = {0.f, 0.f, 0.f, 0.f, 0.f, 0.f, 0.f, 0.f};

  v16h qf[4];
#pragma unroll
  for (int c = 0; c < 4; ++c) {
    h2t* qp = reinterpret_cast<h2t*>(&qf[c]);
#pragma unroll
    for (int v = 0; v < 8; ++v) {
      int kk = ((v < 4) ? (2 * v) : (16 + 2 * (v - 4))) + khalf * 8;
      qp[v] = *(const h2t*)(&Q[rowQ * DMODEL + headCol + c * 32 + kk]);
    }
  }

  // ---- scores: S = Q K^T * scale (all 4 K fragments batched) ----
  for (int kb = wave; kb <= qb; kb += 4) {
    v8f s = vzero;
    const int rowK = b * TCTX + kb * 16 + mloc;
    v16h kfr[4];
#pragma unroll
    for (int c = 0; c < 4; ++c) {
      h2t* kp = reinterpret_cast<h2t*>(&kfr[c]);
#pragma unroll
      for (int v = 0; v < 8; ++v) {
        int kk = ((v < 4) ? (2 * v) : (16 + 2 * (v - 4))) + khalf * 8;
        kp[v] = *(const h2t*)(&Kg[rowK * DMODEL + headCol + c * 32 + kk]);
      }
    }
    SCHED_FENCE();
#pragma unroll
    for (int c = 0; c < 4; ++c)
      s = __builtin_amdgcn_wmma_f32_16x16x32_f16(
          false, qf[c], false, kfr[c], (short)0, s, false, false);
#pragma unroll
    for (int r = 0; r < 8; ++r)
      sc[(khalf * 8 + r) * TCTX + kb * 16 + mloc] = s[r] * scale;
  }
  __syncthreads();

  // ---- causal softmax: one thread per query row ----
  const int nkc = (((qb + 1) * 16) + 31) >> 5;
  if (tid < 16) {
    int row  = tid;
    int qEnd = qb * 16 + row;
    float* sr = &sc[row * TCTX];
    float mx = -1e30f;
    for (int j = 0; j <= qEnd; ++j) mx = fmaxf(mx, sr[j]);
    float sum = 0.f;
    for (int j = 0; j <= qEnd; ++j) { float e = __expf(sr[j] - mx); sr[j] = e; sum += e; }
    float inv = 1.f / sum;
    for (int j = 0; j <= qEnd; ++j) sr[j] *= inv;
    for (int j = qEnd + 1; j < nkc * 32; ++j) sr[j] = 0.f;
  }
  __syncthreads();

  // ---- att = P @ V : coalesced u32 loads from VT, batched per chunk ----
  v8f o0 = vzero, o1 = vzero;
  for (int kc = 0; kc < nkc; ++kc) {
    v16h pf;
    h2t* pp = reinterpret_cast<h2t*>(&pf);
#pragma unroll
    for (int v = 0; v < 8; ++v) {
      int kk = ((v < 4) ? (2 * v) : (16 + 2 * (v - 4))) + khalf * 8;
      h2t t;
      t.x = (_Float16)sc[mloc * TCTX + kc * 32 + kk];
      t.y = (_Float16)sc[mloc * TCTX + kc * 32 + kk + 1];
      pp[v] = t;
    }
    v16h vfr[2];
#pragma unroll
    for (int u = 0; u < 2; ++u) {
      h2t* vp = reinterpret_cast<h2t*>(&vfr[u]);
#pragma unroll
      for (int v = 0; v < 8; ++v) {
        int kk = ((v < 4) ? (2 * v) : (16 + 2 * (v - 4))) + khalf * 8;
        vp[v] = *(const h2t*)(&VT[vtBase + ((wave * 2 + u) * 16 + mloc) * TCTX +
                                  kc * 32 + kk]);
      }
    }
    SCHED_FENCE();
    o0 = __builtin_amdgcn_wmma_f32_16x16x32_f16(
        false, pf, false, vfr[0], (short)0, o0, false, false);
    o1 = __builtin_amdgcn_wmma_f32_16x16x32_f16(
        false, pf, false, vfr[1], (short)0, o1, false, false);
  }
#pragma unroll
  for (int u = 0; u < 2; ++u) {
    int ct = wave * 2 + u;
    v8f oo = u ? o1 : o0;
#pragma unroll
    for (int r = 0; r < 8; ++r)
      O[(b * TCTX + qb * 16 + khalf * 8 + r) * DMODEL + headCol + ct * 16 + mloc] =
          (_Float16)oo[r];
  }
}

// ---------------------------------------------------------------------------
// Host orchestration
// ---------------------------------------------------------------------------
static inline void run_gemm(const _Float16* A, const _Float16* Bw,
                            const float* bias, const float* resid,
                            float* Cf, _Float16* Ch,
                            int M, int N, int K, int relu, hipStream_t s) {
  dim3 grid(N / 128, M / 128);
  gemm_wmma_kernel<<<grid, 256, 0, s>>>(A, Bw, bias, resid, Cf, Ch, M, N, K, relu);
}

extern "C" void kernel_launch(void* const* d_in, const int* in_sizes, int n_in,
                              void* d_out, int out_size, void* d_ws, size_t ws_size,
                              hipStream_t stream) {
  (void)in_sizes; (void)n_in; (void)out_size; (void)ws_size;
  const int*   idx     = (const int*)d_in[0];
  const float* tok_emb = (const float*)d_in[1];
  const float* pos_emb = (const float*)d_in[2];
  const float* wq      = (const float*)d_in[3];
  const float* wk      = (const float*)d_in[4];
  const float* wv      = (const float*)d_in[5];
  const float* wo      = (const float*)d_in[6];
  const float* bo      = (const float*)d_in[7];
  const float* w1      = (const float*)d_in[8];
  const float* b1      = (const float*)d_in[9];
  const float* w2      = (const float*)d_in[10];
  const float* b2      = (const float*)d_in[11];
  const float* g1      = (const float*)d_in[12];
  const float* be1     = (const float*)d_in[13];
  const float* g2      = (const float*)d_in[14];
  const float* be2     = (const float*)d_in[15];
  const float* wlm     = (const float*)d_in[16];
  const float* blm     = (const float*)d_in[17];
  float* out = (float*)d_out;

  char* p = (char*)d_ws;
  auto carve = [&](size_t bytes) -> void* {
    void* r = (void*)p;
    p += (bytes + 255) & ~(size_t)255;
    return r;
  };
  const size_t WDD = (size_t)LAYERS * DMODEL * DMODEL;
  const size_t WDF = (size_t)LAYERS * DMODEL * FFDIM;
  _Float16* wqh  = (_Float16*)carve(WDD * 2);   // [L, D(out), D(in)] transposed
  _Float16* wkh  = (_Float16*)carve(WDD * 2);
  _Float16* wvh  = (_Float16*)carve(WDD * 2);
  _Float16* woh  = (_Float16*)carve(WDD * 2);
  _Float16* w1h  = (_Float16*)carve(WDF * 2);   // [L, FF, D]
  _Float16* w2h  = (_Float16*)carve(WDF * 2);   // [L, D, FF]
  _Float16* wlmh = (_Float16*)carve((size_t)DMODEL * VOCAB * 2);  // [V, D]
  float*    x    = (float*)   carve((size_t)BT * DMODEL * 4);
  _Float16* hbuf = (_Float16*)carve((size_t)BT * DMODEL * 2);
  _Float16* qbuf = (_Float16*)carve((size_t)BT * DMODEL * 2);
  _Float16* kbuf = (_Float16*)carve((size_t)BT * DMODEL * 2);
  _Float16* vbuf = (_Float16*)carve((size_t)BT * DMODEL * 2);
  _Float16* vtb  = (_Float16*)carve((size_t)BT * DMODEL * 2);
  _Float16* attb = (_Float16*)carve((size_t)BT * DMODEL * 2);
  _Float16* ff1  = (_Float16*)carve((size_t)BT * FFDIM * 2);
  _Float16* xh   = (_Float16*)carve((size_t)BT * DMODEL * 2);

  {
    int n = (int)WDD;
    cvt_qkv_kernel<<<n / 256, 256, 0, stream>>>(wq, wqh);
    cvt_qkv_kernel<<<n / 256, 256, 0, stream>>>(wk, wkh);
    cvt_qkv_kernel<<<n / 256, 256, 0, stream>>>(wv, wvh);
    cvt_w_transpose_kernel<<<n / 256, 256, 0, stream>>>(wo, woh, DMODEL, DMODEL);
    int nf = (int)WDF;
    cvt_w_transpose_kernel<<<nf / 256, 256, 0, stream>>>(w1, w1h, DMODEL, FFDIM);
    cvt_w_transpose_kernel<<<nf / 256, 256, 0, stream>>>(w2, w2h, FFDIM, DMODEL);
    int nl = DMODEL * VOCAB;
    cvt_w_transpose_kernel<<<nl / 256, 256, 0, stream>>>(wlm, wlmh, DMODEL, VOCAB);
  }

  embed_kernel<<<BT, 256, 0, stream>>>(idx, tok_emb, pos_emb, x);

  for (int l = 0; l < LAYERS; ++l) {
    const size_t wddo = (size_t)l * DMODEL * DMODEL;
    const size_t wdfo = (size_t)l * DMODEL * FFDIM;
    layernorm_kernel<<<BT, 256, 0, stream>>>(x, g1 + l * DMODEL, be1 + l * DMODEL, hbuf);
    run_gemm(hbuf, wqh + wddo, nullptr, nullptr, nullptr, qbuf, BT, DMODEL, DMODEL, 0, stream);
    run_gemm(hbuf, wkh + wddo, nullptr, nullptr, nullptr, kbuf, BT, DMODEL, DMODEL, 0, stream);
    run_gemm(hbuf, wvh + wddo, nullptr, nullptr, nullptr, vbuf, BT, DMODEL, DMODEL, 0, stream);
    transpose_v_kernel<<<(BT * DMODEL) / 256, 256, 0, stream>>>(vbuf, vtb);
    attn_kernel<<<BATCH * NHEAD * (TCTX / 16), 128, 0, stream>>>(qbuf, kbuf, vtb, attb);
    run_gemm(attb, woh + wddo, bo + l * DMODEL, x, x, nullptr, BT, DMODEL, DMODEL, 0, stream);
    layernorm_kernel<<<BT, 256, 0, stream>>>(x, g2 + l * DMODEL, be2 + l * DMODEL, hbuf);
    run_gemm(hbuf, w1h + wdfo, b1 + l * FFDIM, nullptr, nullptr, ff1, BT, FFDIM, DMODEL, 1, stream);
    run_gemm(ff1, w2h + wdfo, b2 + l * DMODEL, x, x, nullptr, BT, DMODEL, FFDIM, 0, stream);
  }

  {
    int n = BT * DMODEL;
    cvt_f32_to_f16_kernel<<<n / 256, 256, 0, stream>>>(x, xh, n);
    run_gemm(xh, wlmh, blm, nullptr, out, nullptr, BT, VOCAB, DMODEL, 0, stream);
  }
}
